// ASAPNetsHRStream_4947802325655
// MI455X (gfx1250) — matrix operand order repaired and checked by reading, not compiled, not run on hardware
//
#include <hip/hip_runtime.h>

typedef __attribute__((ext_vector_type(16))) _Float16 v16h;
typedef __attribute__((ext_vector_type(8)))  float    v8f;
typedef __attribute__((ext_vector_type(4)))  int      v4i;

#define NIN   13
#define NP    14595          // params per low-res pixel
#define IMG   512
#define HWLR  32             // 32x32 low-res grid
#define AST   72             // activation row stride in f16 (144B, 16B aligned)
#define WST   72             // weight row stride in f16

// LDS layout (bytes):
//   act  : [256][AST] f16   @ 0      (36864 B)
//   wgt  : [288][WST] f16   @ 36864  (41472 B)  rows: L0 0..31, L1 32..95, L2 96..159, L3 160..223, L4 224..287
//   bias : [272]      f32   @ 78336  (1088 B)   L0 0..63, L1 64..127, L2 128..191, L3 192..255, L4 256..258
#define LDS_WGT_OFF  36864
#define LDS_BIAS_OFF 78336
#define LDS_BYTES    79424

__device__ __forceinline__ v8f splat8(float x) {
  v8f v = {x, x, x, x, x, x, x, x};
  return v;
}

// A-fragment (16x32 f16): lane = M%16 + 16*Khalf; elems 0..7 = ch[base+hk*8 ..+8),
// elems 8..15 = ch[base+16+hk*8 ..+8)  -> two 16B LDS loads.
__device__ __forceinline__ v16h loadA(const _Float16* act, int lane, int rowBase, int chBase) {
  const int m = lane & 15, hk = lane >> 4;
  const _Float16* r = act + (rowBase + m) * AST + chBase + hk * 8;
  union { v16h h; v4i q[2]; } u;
  u.q[0] = *(const v4i*)(r);
  u.q[1] = *(const v4i*)(r + 16);
  return u.h;
}

// B-fragment (32x16 f16): lane = N%16 + 16*Khalf; elems 0..7 = K hk*8..+8,
// elems 8..15 = K 16+hk*8..+8 (strided scalar LDS loads).
__device__ __forceinline__ v16h loadB(const _Float16* wgt, int lane, int wrow0, int ks, int nt) {
  const int hk = lane >> 4;
  const _Float16* r = wgt + (wrow0 + ks * 32 + hk * 8) * WST + nt * 16 + (lane & 15);
  v16h h;
#pragma unroll
  for (int s = 0; s < 8; ++s) {
    h[s] = r[s * WST];
    h[s + 8] = r[(16 + s) * WST];
  }
  return h;
}

template <int KSTEPS>
__device__ __forceinline__ void hidden_layer(_Float16* act, const _Float16* wgt,
                                             const float* bias, int lane, int m0,
                                             int wrow0, int bofs) {
  v16h B[4][KSTEPS];
#pragma unroll
  for (int nt = 0; nt < 4; ++nt)
#pragma unroll
    for (int ks = 0; ks < KSTEPS; ++ks)
      B[nt][ks] = loadB(wgt, lane, wrow0, ks, nt);

  const int n = lane & 15, hk = lane >> 4;
#pragma unroll
  for (int mt = 0; mt < 2; ++mt) {
    const int rb = m0 + mt * 16;
    v16h A[KSTEPS];
#pragma unroll
    for (int ks = 0; ks < KSTEPS; ++ks) A[ks] = loadA(act, lane, rb, ks * 32);
    v8f acc[4];
#pragma unroll
    for (int nt = 0; nt < 4; ++nt) acc[nt] = splat8(bias[bofs + nt * 16 + n]);
#pragma unroll
    for (int ks = 0; ks < KSTEPS; ++ks)
#pragma unroll
      for (int nt = 0; nt < 4; ++nt)
        acc[nt] = __builtin_amdgcn_wmma_f32_16x16x32_f16(
            false, A[ks], false, B[nt][ks], (short)0, acc[nt], false, false);
    // leaky ReLU (slope 0.01 > 0  =>  leaky(x) == max(x, 0.01x), no cmp/cndmask)
    // + store f16, in place (this wave's rows only)
#pragma unroll
    for (int nt = 0; nt < 4; ++nt)
#pragma unroll
      for (int r = 0; r < 8; ++r) {
        const float x = acc[nt][r];
        const float y = fmaxf(x, 0.01f * x);
        act[(rb + r + 8 * hk) * AST + nt * 16 + n] = (_Float16)y;
      }
  }
}

__device__ __forceinline__ void final_layer(const _Float16* act, const _Float16* wgt,
                                            const float* bias, float* __restrict__ out,
                                            int lane, int m0, int b, int ylr, int xlr) {
  v16h B0 = loadB(wgt, lane, 224, 0, 0);
  v16h B1 = loadB(wgt, lane, 224, 1, 0);
  const int n = lane & 15, hk = lane >> 4;
  const float bv = bias[256 + n];  // slots 3..15 zeroed
#pragma unroll
  for (int mt = 0; mt < 2; ++mt) {
    const int rb = m0 + mt * 16;
    v16h A0 = loadA(act, lane, rb, 0);
    v16h A1 = loadA(act, lane, rb, 32);
    v8f acc = splat8(bv);
    acc = __builtin_amdgcn_wmma_f32_16x16x32_f16(false, A0, false, B0, (short)0, acc, false, false);
    acc = __builtin_amdgcn_wmma_f32_16x16x32_f16(false, A1, false, B1, (short)0, acc, false, false);
    if (n < 3) {  // divergence only after last WMMA
      const int ly = rb >> 4;  // tile row 0..15
      size_t obase = (((size_t)b * 3 + n) * IMG + (size_t)(ylr * 16 + ly)) * IMG +
                     (size_t)(xlr * 16 + 8 * hk);
#pragma unroll
      for (int r = 0; r < 8; ++r) out[obase + r] = tanhf(acc[r]);
    }
  }
}

__global__ void __launch_bounds__(256) asapnet_mlp_kernel(
    const float* __restrict__ highres, const float* __restrict__ lrp,
    float* __restrict__ out) {
  extern __shared__ char lds_raw[];
  _Float16* act  = (_Float16*)lds_raw;
  _Float16* wgt  = (_Float16*)(lds_raw + LDS_WGT_OFF);
  float*    bias = (float*)(lds_raw + LDS_BIAS_OFF);

  const int t   = threadIdx.x;
  const int blk = blockIdx.x;            // b*1024 + ylr*32 + xlr
  const int b   = blk >> 10;
  const int ylr = (blk >> 5) & 31;
  const int xlr = blk & 31;

  // ---- zero K-padding of layer-0 weights and N-padding of layer-4 bias ----
  for (int i = t; i < 3 * 64; i += 256)
    wgt[(29 + (i >> 6)) * WST + (i & 63)] = (_Float16)0.f;
  if (t >= 3 && t < 16) bias[256 + t] = 0.f;

  // ---- stage per-pixel MLP params: f32 strided gather -> f16 LDS ----
  // flat layout per layer: bias[nco] then weight[nci*nco] (row-major [nci][nco])
  // boundaries: L0 [0,1920) L1 [1920,6080) L2 [6080,10240) L3 [10240,14400) L4 [14400,14595)
  const size_t gbase = (size_t)b * NP * 1024 + (size_t)(ylr * 32 + xlr);
  for (int c = t; c < NP; c += 256) {
    float v = lrp[gbase + (size_t)c * 1024];
    int l, r;
    if (c < 1920)       { l = 0; r = c; }
    else if (c < 6080)  { l = 1; r = c - 1920; }
    else if (c < 10240) { l = 2; r = c - 6080; }
    else if (c < 14400) { l = 3; r = c - 10240; }
    else                { l = 4; r = c - 14400; }
    const int nco   = (l == 4) ? 3 : 64;
    const int wrow0 = (l == 0) ? 0 : 32 + (l - 1) * 64;
    if (r < nco) {
      bias[l * 64 + r] = v;
    } else {
      const int wi = r - nco;
      const int i  = (l == 4) ? (wi / 3) : (wi >> 6);  // input channel (K)
      const int o  = wi - i * nco;                     // output channel (N)
      wgt[(wrow0 + i) * WST + o] = (_Float16)v;
    }
  }

  // ---- stage input features: 13 highres channels + 16 cosine coord channels ----
  {
    const int lx = t & 15, ly = t >> 4;
    const size_t hbase = (size_t)b * NIN * IMG * IMG +
                         (size_t)(ylr * 16 + ly) * IMG + (size_t)(xlr * 16 + lx);
    float f[32];
#pragma unroll
    for (int ch = 0; ch < NIN; ++ch)
      f[ch] = highres[hbase + (size_t)ch * (IMG * IMG)];
    const float TWO_PI = 6.28318530717958647692f;
    int fr = 16;
#pragma unroll
    for (int j = 0; j < 4; ++j) {
      const float px = TWO_PI * (float)(lx & (fr - 1)) / (float)fr;
      const float py = TWO_PI * (float)(ly & (fr - 1)) / (float)fr;
      f[NIN + 4 * j + 0] = __cosf(px);
      f[NIN + 4 * j + 1] = __sinf(px);
      f[NIN + 4 * j + 2] = __cosf(py);
      f[NIN + 4 * j + 3] = __sinf(py);
      fr >>= 1;
    }
    f[29] = 0.f; f[30] = 0.f; f[31] = 0.f;   // K padding
    union { _Float16 h[32]; v4i q[4]; } u;
#pragma unroll
    for (int k = 0; k < 32; ++k) u.h[k] = (_Float16)f[k];
    v4i* dst = (v4i*)(act + t * AST);        // 144B row stride -> 16B aligned
    dst[0] = u.q[0]; dst[1] = u.q[1]; dst[2] = u.q[2]; dst[3] = u.q[3];
  }

  __syncthreads();  // weights are shared cross-wave; activations are per-wave after this

  const int lane = t & 31;
  const int m0   = (t >> 5) * 32;  // each wave owns 32 pixel rows for all layers

  hidden_layer<1>(act, wgt, bias, lane, m0,   0,   0);   // 29(->32) -> 64
  hidden_layer<2>(act, wgt, bias, lane, m0,  32,  64);   // 64 -> 64
  hidden_layer<2>(act, wgt, bias, lane, m0,  96, 128);   // 64 -> 64
  hidden_layer<2>(act, wgt, bias, lane, m0, 160, 192);   // 64 -> 64
  final_layer(act, wgt, bias, out, lane, m0, b, ylr, xlr);  // 64 -> 3, tanh
}

extern "C" void kernel_launch(void* const* d_in, const int* in_sizes, int n_in,
                              void* d_out, int out_size, void* d_ws, size_t ws_size,
                              hipStream_t stream) {
  (void)n_in; (void)out_size; (void)d_ws; (void)ws_size;
  const float* highres = (const float*)d_in[0];
  const float* lrp     = (const float*)d_in[1];
  float* out           = (float*)d_out;

  const int bs = in_sizes[0] / (NIN * IMG * IMG);  // = 4
  dim3 grid(bs * HWLR * HWLR), block(256);
  asapnet_mlp_kernel<<<grid, block, LDS_BYTES, stream>>>(highres, lrp, out);
}
